// guchioGRU1_45956150067333
// MI455X (gfx1250) — matrix operand order repaired and checked by reading, not compiled, not run on hardware
//
#include <hip/hip_runtime.h>
#include <hip/hip_bf16.h>

// ---------------------------------------------------------------------------
// Problem constants (from the reference)
// ---------------------------------------------------------------------------
#define BB    2048
#define LL    107
#define EE    128
#define HH    128
#define D3    384      // 3*H gate dim == padded input dim
#define PRED  68
#define NLAY  9        // NG*NL
#define NROW  (BB*LL)  // 219136, divisible by 64

typedef __attribute__((ext_vector_type(16))) _Float16 v16h;
typedef __attribute__((ext_vector_type(8)))  _Float16 v8h;
typedef __attribute__((ext_vector_type(8)))  float    v8f;

// ---------------------------------------------------------------------------
// WMMA helpers (CDNA5 gfx1250, wave32)
// ---------------------------------------------------------------------------
__device__ __forceinline__ v8f wmma_f16(v16h a, v16h b, v8f c) {
  // D = A(16x32 f16) * B(32x16 f16) + C(16x16 f32)
  return __builtin_amdgcn_wmma_f32_16x16x32_f16(
      /*neg_a=*/false, a, /*neg_b=*/false, b,
      /*c_mod=*/(short)0, c, /*reuse_a=*/false, /*reuse_b=*/false);
}

// A fragment (16x32 f16, row-major source with leading dim `ld`):
//   lane L: M = L&15, hi = L>>4
//   VGPR v holds K = 16*(v>>2) + 8*hi + 2*(v&3) + {0,1}
__device__ __forceinline__ v16h load_a16(const _Float16* base, int ld, int m,
                                         int k0, int hi) {
  union { v16h h; unsigned int u[8]; } a;
  const _Float16* p = base + (size_t)m * ld + k0 + 8 * hi;
#pragma unroll
  for (int v = 0; v < 8; ++v) {
    const int K = ((v >> 2) * 16) + ((v & 3) * 2);
    a.u[v] = *(const unsigned int*)(p + K);
  }
  return a.h;
}

// B fragment (32x16 f16): lane: N = L&15, hi = L>>4; element s holds
// K = 16*(s>>3) + 8*hi + (s&7). Source Wt is (K,N) row-major, ld D3.
__device__ __forceinline__ v16h load_b16(const _Float16* Wt, int k0, int n0,
                                         int lane) {
  v16h b;
  const _Float16* p = Wt + (size_t)(k0 + 8 * (lane >> 4)) * D3 + n0 + (lane & 15);
#pragma unroll
  for (int s = 0; s < 16; ++s)
    b[s] = p[(size_t)(((s >> 3) * 16) + (s & 7)) * D3];
  return b;
}

// Async global -> LDS 16-byte copy (CDNA5 ASYNCcnt path).
__device__ __forceinline__ void async_g2l_b128(void* lds_dst,
                                               const void* gsrc) {
  const unsigned lds = (unsigned)(uintptr_t)lds_dst;
  const unsigned long long ga = (unsigned long long)(uintptr_t)gsrc;
  asm volatile("global_load_async_to_lds_b128 %0, %1, off"
               :: "v"(lds), "v"(ga)
               : "memory");
}
__device__ __forceinline__ void wait_async_le3() {
  asm volatile("s_wait_asynccnt 0x3" ::: "memory");
}
__device__ __forceinline__ void wait_async_0() {
  asm volatile("s_wait_asynccnt 0x0" ::: "memory");
}

__device__ __forceinline__ float fast_sigmoid(float x) {
  return 1.0f / (1.0f + __expf(-x));
}
__device__ __forceinline__ float fast_tanh(float x) {
  const float e = __expf(2.0f * x);
  return (e - 1.0f) / (e + 1.0f);
}

// ---------------------------------------------------------------------------
// Kernel 1: embedding gather -> f16 activations (B*L, 384)
// ---------------------------------------------------------------------------
__global__ void embed_kernel(const int* __restrict__ seq,
                             const int* __restrict__ str,
                             const int* __restrict__ lp,
                             const float* __restrict__ tbl,
                             _Float16* __restrict__ X) {
  const int row = blockIdx.x;       // b*L + l
  const int e   = threadIdx.x;      // 0..127
  _Float16* o = X + (size_t)row * D3;
  o[e]        = (_Float16)tbl[seq[row] * EE + e];
  o[EE + e]   = (_Float16)tbl[str[row] * EE + e];
  o[2*EE + e] = (_Float16)tbl[lp[row]  * EE + e];
}

// ---------------------------------------------------------------------------
// Kernel 2: weight convert+transpose to f16 (K,N)-major for WMMA B fragments
// ---------------------------------------------------------------------------
__global__ void wconv_kernel(const float* __restrict__ w_ih,
                             const float* __restrict__ w_hh,
                             _Float16* __restrict__ WihT,
                             _Float16* __restrict__ WhhT) {
  const long total1 = 18L * 384 * 384;
  const long total2 = 18L * 384 * 128;
  for (long idx = (long)blockIdx.x * blockDim.x + threadIdx.x;
       idx < total1 + total2; idx += (long)gridDim.x * blockDim.x) {
    if (idx < total1) {
      const long dir = idx / (384 * 384);
      const long r   = idx % (384 * 384);
      const long n = r / 384, k = r % 384;
      WihT[dir * 384 * 384 + k * 384 + n] = (_Float16)w_ih[idx];
    } else {
      const long j = idx - total1;
      const long dir = j / (384 * 128);
      const long r   = j % (384 * 128);
      const long n = r / 128, k = r % 128;
      WhhT[dir * 128 * 384 + k * 384 + n] = (_Float16)w_hh[j];
    }
  }
}

// ---------------------------------------------------------------------------
// Kernel 3: input projection GEMM, fused fwd/bwd via blockIdx.y.
//   XP[m][n] = sum_k X[m][k] * WihT[k][n] + bih[n]   (M=NROW, N=384, K=384)
// blockIdx.x = 64-row M block: the contiguous 48KB A block is staged once via
// async-to-LDS; 8 waves split the 24 N tiles (3 each) and iterate 4 M tiles,
// so each B fragment read from L2 feeds 4 WMMAs (4x weight-traffic reuse) and
// X is read exactly once from HBM.
// ---------------------------------------------------------------------------
__global__ void __launch_bounds__(256)
xp_gemm_kernel(const _Float16* __restrict__ X,
               const _Float16* __restrict__ WihT2,  // 2 dirs back-to-back
               const float* __restrict__ bih2,      // 2 dirs back-to-back
               _Float16* __restrict__ XPf,
               _Float16* __restrict__ XPb) {
  __shared__ _Float16 sA[64 * D3];   // 48KB A block (4 x 16-row tiles)

  const int d    = blockIdx.y;
  const _Float16* Wt   = WihT2 + (size_t)d * 384 * 384;
  const float*    bias = bih2 + d * 384;
  _Float16*       XP   = d ? XPb : XPf;

  const int tid  = threadIdx.x;
  const int wave = tid >> 5;          // = N group (8 groups x 3 tiles = 24)
  const int lane = tid & 31;
  const int hi   = lane >> 4;
  const int ml   = lane & 15;
  const int mb   = blockIdx.x;        // 64-row block

  // Stage the contiguous 49152-byte A block: 3072 x b128, 12 per thread.
  {
    const _Float16* g = X + (size_t)mb * 64 * D3;
#pragma unroll
    for (int c = 0; c < 12; ++c) {
      const int q = c * 256 + tid;            // 16B chunk index
      async_g2l_b128(&sA[q * 8], g + q * 8);
    }
  }
  wait_async_0();
  __syncthreads();

  v8f acc[4][3] = {};
  for (int k0 = 0; k0 < 384; k0 += 32) {
    v16h a[4];
#pragma unroll
    for (int mt = 0; mt < 4; ++mt)
      a[mt] = load_a16(sA + mt * 16 * D3, D3, ml, k0, hi);
#pragma unroll
    for (int i = 0; i < 3; ++i) {
      const v16h b = load_b16(Wt, k0, wave * 48 + i * 16, lane);
#pragma unroll
      for (int mt = 0; mt < 4; ++mt)
        acc[mt][i] = wmma_f16(a[mt], b, acc[mt][i]);
    }
  }

#pragma unroll
  for (int i = 0; i < 3; ++i) {
    const int n = wave * 48 + i * 16 + ml;
    const float bv = bias[n];
#pragma unroll
    for (int mt = 0; mt < 4; ++mt) {
#pragma unroll
      for (int r = 0; r < 8; ++r) {
        const int row = mb * 64 + mt * 16 + hi * 8 + r;
        XP[(size_t)row * D3 + n] = (_Float16)(acc[mt][i][r] + bv);
      }
    }
  }
}

// ---------------------------------------------------------------------------
// Kernel 4: persistent bidirectional GRU scan.
//   grid (128, 2): x = 16-row batch tile, y = direction (1 == reverse).
//   Whh^T lives in registers as 12 B-fragments per wave for all 107 steps.
//   XP for step t+1 is async-prefetched into a double-buffered LDS tile
//   while step t computes (ASYNCcnt in-order completion => wait <= 3 releases
//   exactly the previous step's tile). Gate phase is fully vectorized:
//   each thread owns 8 consecutive hidden units (16B/32B aligned packets).
// ---------------------------------------------------------------------------
__global__ void __launch_bounds__(256)
gru_scan_kernel(const _Float16* __restrict__ XPf,
                const _Float16* __restrict__ XPb,
                const _Float16* __restrict__ WhhT2,
                const float* __restrict__ bhh2,
                _Float16* __restrict__ Xnext) {
  __shared__ _Float16 sh16[16 * HH];     // h, f16 mirror (A operand)
  __shared__ float    sh32[16 * HH];     // h, f32 master
  __shared__ float    sgh[16 * D3];      // h @ Whh^T result
  __shared__ float    sbias[D3];         // bhh
  __shared__ _Float16 sxp[2][16 * D3];   // double-buffered XP tile (2x12KB)

  const int d = blockIdx.y;
  const _Float16* WhhT = WhhT2 + (size_t)d * 128 * 384;
  const float*    bhh  = bhh2 + d * 384;
  const _Float16* XP   = d ? XPb : XPf;
  const int outOff = d * HH;

  const int tid  = threadIdx.x;
  const int wave = tid >> 5;
  const int lane = tid & 31;
  const int hi   = lane >> 4;
  const int ml   = lane & 15;
  const int b0   = blockIdx.x * 16;

  // Preload Whh^T B-fragments: 3 N-tiles x 4 K-chunks, resident in VGPRs.
  v16h Bf[12];
#pragma unroll
  for (int i = 0; i < 3; ++i)
#pragma unroll
    for (int kc = 0; kc < 4; ++kc)
      Bf[i * 4 + kc] = load_b16(WhhT, kc * 32, wave * 48 + i * 16, lane);

  for (int i = tid; i < 16 * HH; i += 256) {
    sh32[i] = 0.0f;
    sh16[i] = (_Float16)0.0f;
  }
  for (int i = tid; i < D3; i += 256) sbias[i] = bhh[i];

  // Prefetch XP tile for t = 0 (16 rows x 768B, 3 x b128 per thread).
  {
    const int tt0 = d ? (LL - 1) : 0;
#pragma unroll
    for (int c = 0; c < 3; ++c) {
      const int q = c * 256 + tid;
      const int m = q / 48, r = q % 48;   // 48 x 16B chunks per row
      async_g2l_b128(&sxp[0][m * D3 + r * 8],
                     XP + ((size_t)(b0 + m) * LL + tt0) * D3 + r * 8);
    }
  }
  __syncthreads();

  const int gm = tid >> 4;             // batch row within tile for gate phase
  const int jb = (tid & 15) * 8;       // 8 hidden units per thread

  for (int t = 0; t < LL; ++t) {
    const int tt = d ? (LL - 1 - t) : t;

    // Issue async prefetch of next step's XP tile into the other buffer.
    const bool more = (t + 1 < LL);
    if (more) {
      const int ttn = d ? (LL - 2 - t) : (t + 1);
      _Float16* dst = sxp[(t + 1) & 1];
#pragma unroll
      for (int c = 0; c < 3; ++c) {
        const int q = c * 256 + tid;
        const int m = q / 48, r = q % 48;
        async_g2l_b128(&dst[m * D3 + r * 8],
                       XP + ((size_t)(b0 + m) * LL + ttn) * D3 + r * 8);
      }
    }

    // ---- GEMM phase: gh = h @ Whh^T (12 wmma / wave) ----
    v8f acc[3] = {};
#pragma unroll
    for (int kc = 0; kc < 4; ++kc) {
      const v16h a = load_a16(sh16, HH, ml, kc * 32, hi);
#pragma unroll
      for (int i = 0; i < 3; ++i) acc[i] = wmma_f16(a, Bf[i * 4 + kc], acc[i]);
    }
#pragma unroll
    for (int i = 0; i < 3; ++i) {
      const int n = wave * 48 + i * 16 + ml;
#pragma unroll
      for (int r = 0; r < 8; ++r) sgh[(hi * 8 + r) * D3 + n] = acc[i][r];
    }

    // Current step's tile (issued last iteration) must be complete; async
    // loads finish in order, so <=3 releases the older triple.
    if (more) wait_async_le3();
    else      wait_async_0();
    __syncthreads();

    // ---- Gate phase (vectorized: 8 consecutive units per thread) ----
    const _Float16* xp = sxp[t & 1] + gm * D3 + jb;
    const float*    gh = sgh + gm * D3 + jb;
    const v8f ghr = *(const v8f*)(gh);
    const v8f ghz = *(const v8f*)(gh + HH);
    const v8f ghn = *(const v8f*)(gh + 2 * HH);
    const v8f bsr = *(const v8f*)(sbias + jb);
    const v8f bsz = *(const v8f*)(sbias + HH + jb);
    const v8f bsn = *(const v8f*)(sbias + 2 * HH + jb);
    const v8h xr8 = *(const v8h*)(xp);
    const v8h xz8 = *(const v8h*)(xp + HH);
    const v8h xn8 = *(const v8h*)(xp + 2 * HH);
    const v8f hp8 = *(const v8f*)(sh32 + gm * HH + jb);

    v8f hv8;
    v8h hv8h;
#pragma unroll
    for (int jj = 0; jj < 8; ++jj) {
      const float rg = fast_sigmoid((float)xr8[jj] + ghr[jj] + bsr[jj]);
      const float zg = fast_sigmoid((float)xz8[jj] + ghz[jj] + bsz[jj]);
      const float ng = fast_tanh((float)xn8[jj] + rg * (ghn[jj] + bsn[jj]));
      const float hv = (1.0f - zg) * ng + zg * hp8[jj];
      hv8[jj]  = hv;
      hv8h[jj] = (_Float16)hv;
    }
    *(v8f*)(sh32 + gm * HH + jb) = hv8;
    *(v8h*)(sh16 + gm * HH + jb) = hv8h;

    const size_t rowoff = ((size_t)(b0 + gm) * LL + tt) * D3;
    *(v8h*)(Xnext + rowoff + outOff + jb) = hv8h;
    if (d == 0) {
      const v8h z8 = {};
      *(v8h*)(Xnext + rowoff + 2 * HH + jb) = z8;   // zero pad tail
    }
    __syncthreads();
  }
}

// ---------------------------------------------------------------------------
// Kernel 5: head — out[b,p,:] = Xlast[b,p,0:256] @ lin_w + lin_b
// ---------------------------------------------------------------------------
__global__ void head_kernel(const _Float16* __restrict__ Xlast,
                            const float* __restrict__ w,
                            const float* __restrict__ bv,
                            float* __restrict__ out) {
  const int idx = blockIdx.x * blockDim.x + threadIdx.x;
  if (idx >= BB * PRED) return;
  const int b = idx / PRED, p = idx % PRED;
  const _Float16* x = Xlast + ((size_t)b * LL + p) * D3;
  float s0 = bv[0], s1 = bv[1], s2 = bv[2];
  for (int j = 0; j < 2 * HH; ++j) {
    const float xv = (float)x[j];
    s0 += xv * w[j * 3 + 0];
    s1 += xv * w[j * 3 + 1];
    s2 += xv * w[j * 3 + 2];
  }
  out[idx * 3 + 0] = s0;
  out[idx * 3 + 1] = s1;
  out[idx * 3 + 2] = s2;
}

// ---------------------------------------------------------------------------
// Host launcher
// ---------------------------------------------------------------------------
extern "C" void kernel_launch(void* const* d_in, const int* in_sizes, int n_in,
                              void* d_out, int out_size, void* d_ws,
                              size_t ws_size, hipStream_t stream) {
  const int*   seq   = (const int*)d_in[0];
  const int*   str   = (const int*)d_in[1];
  const int*   lp    = (const int*)d_in[2];
  const float* tbl   = (const float*)d_in[3];
  const float* w_ih  = (const float*)d_in[4];
  const float* w_hh  = (const float*)d_in[5];
  const float* b_ih  = (const float*)d_in[6];
  const float* b_hh  = (const float*)d_in[7];
  const float* lin_w = (const float*)d_in[8];
  const float* lin_b = (const float*)d_in[9];
  float* out = (float*)d_out;

  // Workspace carve-up (f16 halves):
  _Float16* WihT = (_Float16*)d_ws;
  _Float16* WhhT = WihT + (size_t)18 * 384 * 384;
  _Float16* Xa   = WhhT + (size_t)18 * 128 * 384;
  _Float16* Xb   = Xa + (size_t)NROW * D3;
  _Float16* XPf  = Xb + (size_t)NROW * D3;
  _Float16* XPb  = XPf + (size_t)NROW * D3;

  embed_kernel<<<NROW, 128, 0, stream>>>(seq, str, lp, tbl, Xa);
  wconv_kernel<<<4096, 256, 0, stream>>>(w_ih, w_hh, WihT, WhhT);

  _Float16* xcur = Xa;
  _Float16* xnext = Xb;
  for (int layer = 0; layer < NLAY; ++layer) {
    const size_t wih_off  = (size_t)layer * 2 * 384 * 384;
    const size_t whh_off  = (size_t)layer * 2 * 128 * 384;
    const size_t bias_off = (size_t)layer * 2 * 384;

    dim3 ggrid(NROW / 64, 2);
    xp_gemm_kernel<<<ggrid, 256, 0, stream>>>(
        xcur, WihT + wih_off, b_ih + bias_off, XPf, XPb);

    dim3 sgrid(BB / 16, 2);
    gru_scan_kernel<<<sgrid, 256, 0, stream>>>(
        XPf, XPb, WhhT + whh_off, b_hh + bias_off, xnext);

    _Float16* tmp = xcur; xcur = xnext; xnext = tmp;
  }

  head_kernel<<<(BB * PRED + 255) / 256, 256, 0, stream>>>(xcur, lin_w, lin_b,
                                                           out);
}
